// QwenAttention_89988154785927
// MI455X (gfx1250) — matrix operand order repaired and checked by reading, not compiled, not run on hardware
//
#include <hip/hip_runtime.h>
#include <hip/hip_bf16.h>

// Problem constants (match reference)
#define B_  2
#define S_  2048
#define D_  2048
#define H_  16
#define KV_ 4
#define HD_ 128
#define GR_ (H_ / KV_)      // GQA group size = 4
#define M_  (B_ * S_)       // 4096 rows for the projection GEMMs

typedef __bf16 bf16;
typedef __bf16 v16bf __attribute__((ext_vector_type(16)));
typedef __bf16 v8bf  __attribute__((ext_vector_type(8)));
typedef float  v8f   __attribute__((ext_vector_type(8)));
typedef unsigned int u32x4 __attribute__((ext_vector_type(4)));
typedef int          i32x8 __attribute__((ext_vector_type(8)));
typedef int          i32x4 __attribute__((ext_vector_type(4)));

static __device__ __forceinline__ v16bf join8(v8bf a, v8bf b) {
    return __builtin_shufflevector(a, b, 0,1,2,3,4,5,6,7,8,9,10,11,12,13,14,15);
}

// ---------------------------------------------------------------------------
// fp32 -> bf16 elementwise convert (4 elems/thread, b128 loads)
// ---------------------------------------------------------------------------
__global__ void k_cvt_bf16(const float* __restrict__ in, bf16* __restrict__ out, int n) {
    int i = (blockIdx.x * blockDim.x + threadIdx.x) * 4;
    if (i + 3 < n) {
        float4 f = *(const float4*)(in + i);
        out[i + 0] = (bf16)f.x;
        out[i + 1] = (bf16)f.y;
        out[i + 2] = (bf16)f.z;
        out[i + 3] = (bf16)f.w;
    } else {
        for (; i < n; ++i) out[i] = (bf16)in[i];
    }
}

// ---------------------------------------------------------------------------
// Weight convert + transpose: W[K][N] fp32 -> Wt[N][K] bf16
// ---------------------------------------------------------------------------
__global__ void k_wtrans(const float* __restrict__ W, bf16* __restrict__ Wt, int K, int N) {
    int idx = blockIdx.x * blockDim.x + threadIdx.x;
    if (idx >= K * N) return;
    int k = idx / N;
    int n = idx % N;
    Wt[(size_t)n * K + k] = (bf16)W[idx];
}

// ---------------------------------------------------------------------------
// bf16 WMMA GEMM: out[M][N] = A[M][K] @ Bt[N][K]^T (+ bias), fp32 accumulate.
// Block = 256 threads = 8 waves arranged 4 (M) x 2 (N); block tile 128x128.
// Each wave computes a 32x64 macro-tile = 2x4 WMMA accumulators so the two
// A-fragments are reused across 4 N-subtiles and each B-fragment across 2
// M-subtiles: 12 b128 loads per 8 WMMAs instead of 4 loads per WMMA.
// Fragment layouts per CDNA5 ISA tables:
//   A 16x32 bf16: lane m=lane&15 = row; hi=lane>>4 selects K halves
//     {hi*8..hi*8+7} and {16+hi*8..23+hi*8}.
//   B 32x16 bf16: lane&15 = column; same K halves as A.
//   C/D f32:      VGPR r <-> row r + hi*8; lane&15 <-> column.
// ---------------------------------------------------------------------------
__global__ __launch_bounds__(256) void k_gemm(
        const bf16* __restrict__ A, const bf16* __restrict__ Bt,
        const float* __restrict__ bias,
        bf16* __restrict__ outH, float* __restrict__ outF,
        int N, int K) {
    const int wave  = threadIdx.x >> 5;
    const int lane  = threadIdx.x & 31;
    const int hi    = lane >> 4;
    const int ln    = lane & 15;
    const int waveN = wave & 1;                      // 0..1
    const int waveM = wave >> 1;                     // 0..3
    const int mBase = blockIdx.y * 128 + waveM * 32;
    const int nBase = blockIdx.x * 128 + waveN * 64;

    const bf16* aRow0 = A  + (size_t)(mBase + ln) * K;
    const bf16* aRow1 = A  + (size_t)(mBase + 16 + ln) * K;
    const bf16* bRow0 = Bt + (size_t)(nBase + ln) * K;

    v8f acc[2][4] = {};
    for (int k = 0; k < K; k += 32) {
        __builtin_prefetch(aRow0 + k + 512, 0, 0);   // global_prefetch_b8
        __builtin_prefetch(bRow0 + k + 512, 0, 0);
        v16bf a0 = join8(*(const v8bf*)(aRow0 + k + hi * 8),
                         *(const v8bf*)(aRow0 + k + 16 + hi * 8));
        v16bf a1 = join8(*(const v8bf*)(aRow1 + k + hi * 8),
                         *(const v8bf*)(aRow1 + k + 16 + hi * 8));
#pragma unroll
        for (int j = 0; j < 4; ++j) {
            v16bf bv = join8(*(const v8bf*)(bRow0 + (size_t)j * 16 * K + k + hi * 8),
                             *(const v8bf*)(bRow0 + (size_t)j * 16 * K + k + 16 + hi * 8));
            acc[0][j] = __builtin_amdgcn_wmma_f32_16x16x32_bf16(
                    false, a0, false, bv, (short)0, acc[0][j], false, false);
            acc[1][j] = __builtin_amdgcn_wmma_f32_16x16x32_bf16(
                    false, a1, false, bv, (short)0, acc[1][j], false, false);
        }
    }

#pragma unroll
    for (int i = 0; i < 2; ++i) {
#pragma unroll
        for (int j = 0; j < 4; ++j) {
            const int col = nBase + j * 16 + ln;
            const float badd = bias ? bias[col] : 0.0f;
#pragma unroll
            for (int r = 0; r < 8; ++r) {
                int   row = mBase + i * 16 + hi * 8 + r;
                float v   = acc[i][j][r] + badd;
                if (outF) outF[(size_t)row * N + col] = v;
                else      outH[(size_t)row * N + col] = (bf16)v;
            }
        }
    }
}

// ---------------------------------------------------------------------------
// RoPE: lin (B,S,nh*HD) bf16 -> out (B,nh,S,HD) bf16 with rotary embedding.
// ---------------------------------------------------------------------------
__global__ void k_rope(const bf16* __restrict__ lin, bf16* __restrict__ out, int nh) {
    const int s = blockIdx.x, h = blockIdx.y, b = blockIdx.z;
    const int d = threadIdx.x;                       // 0..63
    const bf16* src = lin + ((size_t)(b * S_ + s) * nh + h) * HD_;
    bf16*       dst = out + ((size_t)(b * nh + h) * S_ + s) * HD_;

    float x1 = (float)src[d];
    float x2 = (float)src[d + 64];
    float invf = __powf(10000.0f, -(float)(2 * d) / (float)HD_);
    float fr = (float)s * invf;
    float c  = __cosf(fr);
    float sn = __sinf(fr);
    dst[d]      = (bf16)(x1 * c - x2 * sn);
    dst[d + 64] = (bf16)(x2 * c + x1 * sn);
}

// ---------------------------------------------------------------------------
// V transpose: vlin (B,S,KV*HD) -> vt (B,KV,HD,S)
// ---------------------------------------------------------------------------
__global__ void k_prepv(const bf16* __restrict__ vlin, bf16* __restrict__ vt) {
    int idx = blockIdx.x * blockDim.x + threadIdx.x;
    int d  = idx % HD_;
    int kv = (idx / HD_) % KV_;
    int s  = (idx / (HD_ * KV_)) % S_;
    int b  = idx / (HD_ * KV_ * S_);
    vt[(((size_t)b * KV_ + kv) * HD_ + d) * S_ + s] = vlin[idx];
}

// ---------------------------------------------------------------------------
// Fused attention per (b, h, 16-query tile). 256 threads = 8 waves.
// Phase 0: TDM (tensor_load_to_lds) DMAs the 16x128 bf16 Q tile into LDS;
//          s_wait_tensorcnt + barrier, then all waves read Q fragments via DS.
// Phase 1: scores 16 x 2048 (fp32) -> LDS via WMMA (Q fragments reused).
// Phase 2: row softmax, wave32 shfl_xor 16-lane reductions, probs bf16 -> LDS.
// Phase 3: P@V via WMMA (A from LDS, B from V^T).
// Dynamic LDS: 128K (scores f32) + 64K (probs bf16) + 4K (Q) = 196 KB < 320 KB.
// ---------------------------------------------------------------------------
__global__ __launch_bounds__(256) void k_attn(
        const bf16* __restrict__ Qr, const bf16* __restrict__ Kr,
        const bf16* __restrict__ Vt, bf16* __restrict__ attn) {
    extern __shared__ __align__(16) char smem[];
    float* sS = (float*)smem;                                        // [16][S_]
    bf16*  sP = (bf16*)(smem + (size_t)16 * S_ * sizeof(float));     // [16][S_]
    bf16*  sQ = (bf16*)(smem + (size_t)16 * S_ * sizeof(float)
                             + (size_t)16 * S_ * sizeof(bf16));      // [16][HD_]

    const int qt = blockIdx.x, h = blockIdx.y, b = blockIdx.z;
    const int kvh  = h / GR_;
    const int wave = threadIdx.x >> 5;
    const int lane = threadIdx.x & 31;
    const int hi   = lane >> 4;
    const int ln   = lane & 15;

    const bf16* Q  = Qr + (((size_t)b * H_ + h) * S_ + qt * 16) * HD_;
    const bf16* Kb = Kr + ((size_t)b * KV_ + kvh) * S_ * HD_;

    // ---- Phase 0: stage Q tile (16x128 bf16, 4KB) into LDS via the TDM ----
#if __has_builtin(__builtin_amdgcn_tensor_load_to_lds) && __has_builtin(__builtin_amdgcn_s_wait_tensorcnt)
    if (wave == 0) {
        unsigned ldsOff = (unsigned)(unsigned long long)(void*)sQ;   // LDS byte offset
        unsigned long long ga = (unsigned long long)(const void*)Q;
        // D# group 0: count=1 | lds_addr | global_addr[56:0] | type=2
        u32x4 g0;
        g0[0] = 1u;                                          // count=1 (user desc)
        g0[1] = ldsOff;                                      // bits 63:32  lds_addr
        g0[2] = (unsigned)(ga & 0xFFFFFFFFu);                // bits 95:64  gaddr lo
        g0[3] = (unsigned)((ga >> 32) & 0x01FFFFFFu)         // bits 120:96 gaddr hi
              | (2u << 30);                                  // bits 127:126 type=2
        // D# group 1: data_size=2B, tensor 128x16, tile 128x16, dim0 stride 128
        i32x8 g1;
        g1[0] = (1 << 16);                                   // data_size=1 -> 2 bytes
        g1[1] = (HD_ & 0xFFFF) << 16;                        // tensor_dim0 = 128
        g1[2] = (16 << 16);                                  // tensor_dim1 = 16
        g1[3] = (HD_ << 16);                                 // tile_dim0   = 128
        g1[4] = 16;                                          // tile_dim1=16, tile_dim2=0
        g1[5] = HD_;                                         // tensor_dim0_stride = 128
        g1[6] = 0;
        g1[7] = 0;
        i32x4 z4 = {0, 0, 0, 0};
        i32x8 z8 = {0, 0, 0, 0, 0, 0, 0, 0};
        // 6-arg form (clang-23 / amdgpu-toolchain): g0, g1, g2, g3, extra, cpol
        __builtin_amdgcn_tensor_load_to_lds(g0, g1, z4, z4, z8, 0);
        __builtin_amdgcn_s_wait_tensorcnt(0);
    }
#else
    for (int i = threadIdx.x; i < 16 * HD_; i += 256) sQ[i] = Q[i];
#endif
    __syncthreads();

    // Q fragments (HD=128 => 4 fragments of K=32) from LDS, reused all phase 1.
    v16bf qa[4];
#pragma unroll
    for (int kk = 0; kk < 4; ++kk) {
        qa[kk] = join8(*(const v8bf*)(sQ + ln * HD_ + kk * 32 + hi * 8),
                       *(const v8bf*)(sQ + ln * HD_ + kk * 32 + 16 + hi * 8));
    }

    // ---- Phase 1: all 128 key tiles, 16 per wave ----
    for (int kt = wave; kt < S_ / 16; kt += 8) {
        const bf16* Krow = Kb + (size_t)kt * 16 * HD_;
        v8f acc = {};
#pragma unroll
        for (int kk = 0; kk < 4; ++kk) {
            v16bf bv = join8(*(const v8bf*)(Krow + ln * HD_ + kk * 32 + hi * 8),
                             *(const v8bf*)(Krow + ln * HD_ + kk * 32 + 16 + hi * 8));
            acc = __builtin_amdgcn_wmma_f32_16x16x32_bf16(
                    false, qa[kk], false, bv, (short)0, acc, false, false);
        }
#pragma unroll
        for (int r = 0; r < 8; ++r)
            sS[(hi * 8 + r) * S_ + kt * 16 + ln] = acc[r];
    }
    __syncthreads();

    // ---- Phase 2: softmax over each of the 16 rows (16 threads per row) ----
    {
        const float scale = 0.08838834764831845f;   // 1/sqrt(128)
        int row = threadIdx.x >> 4;
        int j   = threadIdx.x & 15;
        float* rowp = sS + (size_t)row * S_;

        float mx = -3.4e38f;
        for (int k = j; k < S_; k += 16) mx = fmaxf(mx, rowp[k]);
        for (int o = 1; o < 16; o <<= 1) mx = fmaxf(mx, __shfl_xor(mx, o, 32));
        mx *= scale;

        float sum = 0.0f;
        for (int k = j; k < S_; k += 16) sum += __expf(rowp[k] * scale - mx);
        for (int o = 1; o < 16; o <<= 1) sum += __shfl_xor(sum, o, 32);
        float inv = 1.0f / sum;

        bf16* prow = sP + (size_t)row * S_;
        for (int k = j; k < S_; k += 16)
            prow[k] = (bf16)(__expf(rowp[k] * scale - mx) * inv);
    }
    __syncthreads();

    // ---- Phase 3: out tile = P(16 x 2048) @ V(2048 x 16); wave owns 16 dims ----
    const bf16* Vrow = Vt + (((size_t)b * KV_ + kvh) * HD_ + wave * 16) * S_;
    v8f acc = {};
    for (int kb = 0; kb < S_; kb += 32) {
        v16bf pa = join8(*(const v8bf*)(sP + (size_t)ln * S_ + kb + hi * 8),
                         *(const v8bf*)(sP + (size_t)ln * S_ + kb + 16 + hi * 8));
        v16bf vv = join8(*(const v8bf*)(Vrow + (size_t)ln * S_ + kb + hi * 8),
                         *(const v8bf*)(Vrow + (size_t)ln * S_ + kb + 16 + hi * 8));
        acc = __builtin_amdgcn_wmma_f32_16x16x32_bf16(
                false, pa, false, vv, (short)0, acc, false, false);
    }
#pragma unroll
    for (int r = 0; r < 8; ++r) {
        int srow = qt * 16 + hi * 8 + r;
        attn[((size_t)(b * S_ + srow)) * (H_ * HD_) + h * HD_ + wave * 16 + ln] =
            (bf16)acc[r];
    }
}

// ---------------------------------------------------------------------------
// Host orchestration
// ---------------------------------------------------------------------------
extern "C" void kernel_launch(void* const* d_in, const int* in_sizes, int n_in,
                              void* d_out, int out_size, void* d_ws, size_t ws_size,
                              hipStream_t stream) {
    (void)in_sizes; (void)n_in; (void)out_size; (void)ws_size;

    const float* hidden = (const float*)d_in[0];
    const float* Wq = (const float*)d_in[1];
    const float* bq = (const float*)d_in[2];
    const float* Wk = (const float*)d_in[3];
    const float* bk = (const float*)d_in[4];
    const float* Wv = (const float*)d_in[5];
    const float* bv = (const float*)d_in[6];
    const float* Wo = (const float*)d_in[7];
    float* out = (float*)d_out;

    char* ws = (char*)d_ws;
    size_t off = 0;
    auto carve = [&](size_t elems) -> bf16* {
        bf16* p = (bf16*)(ws + off);
        off += (elems * sizeof(bf16) + 255) & ~(size_t)255;
        return p;
    };
    bf16* hiddenB = carve((size_t)M_ * D_);
    bf16* WqT   = carve((size_t)(H_ * HD_) * D_);
    bf16* WkT   = carve((size_t)(KV_ * HD_) * D_);
    bf16* WvT   = carve((size_t)(KV_ * HD_) * D_);
    bf16* WoT   = carve((size_t)D_ * (H_ * HD_));
    bf16* qLin  = carve((size_t)M_ * H_ * HD_);
    bf16* kLin  = carve((size_t)M_ * KV_ * HD_);
    bf16* vLin  = carve((size_t)M_ * KV_ * HD_);
    bf16* qRope = carve((size_t)B_ * H_ * S_ * HD_);
    bf16* kRope = carve((size_t)B_ * KV_ * S_ * HD_);
    bf16* vT    = carve((size_t)B_ * KV_ * HD_ * S_);
    bf16* attnB = carve((size_t)M_ * H_ * HD_);

    // 1. Convert activations & weights to bf16 (weights transposed to N-major).
    { int n = M_ * D_; k_cvt_bf16<<<n / 1024, 256, 0, stream>>>(hidden, hiddenB, n); }
    { int n = D_ * H_ * HD_;  k_wtrans<<<(n + 255) / 256, 256, 0, stream>>>(Wq, WqT, D_, H_ * HD_); }
    { int n = D_ * KV_ * HD_; k_wtrans<<<(n + 255) / 256, 256, 0, stream>>>(Wk, WkT, D_, KV_ * HD_); }
    { int n = D_ * KV_ * HD_; k_wtrans<<<(n + 255) / 256, 256, 0, stream>>>(Wv, WvT, D_, KV_ * HD_); }
    { int n = (H_ * HD_) * D_; k_wtrans<<<(n + 255) / 256, 256, 0, stream>>>(Wo, WoT, H_ * HD_, D_); }

    dim3 blk(256);
    // 2. QKV projections (bf16 WMMA, +bias, bf16 out). Block tile 128x128.
    k_gemm<<<dim3((H_ * HD_) / 128,  M_ / 128), blk, 0, stream>>>(hiddenB, WqT, bq, qLin, nullptr, H_ * HD_,  D_);
    k_gemm<<<dim3((KV_ * HD_) / 128, M_ / 128), blk, 0, stream>>>(hiddenB, WkT, bk, kLin, nullptr, KV_ * HD_, D_);
    k_gemm<<<dim3((KV_ * HD_) / 128, M_ / 128), blk, 0, stream>>>(hiddenB, WvT, bv, vLin, nullptr, KV_ * HD_, D_);

    // 3. RoPE + V transpose.
    k_rope<<<dim3(S_, H_, B_),  64, 0, stream>>>(qLin, qRope, H_);
    k_rope<<<dim3(S_, KV_, B_), 64, 0, stream>>>(kLin, kRope, KV_);
    k_prepv<<<(M_ * KV_ * HD_) / 256, 256, 0, stream>>>(vLin, vT);

    // 4. Fused attention (TDM Q-stage -> scores -> softmax -> P@V).
    size_t smem = (size_t)16 * S_ * sizeof(float)
                + (size_t)16 * S_ * sizeof(bf16)
                + (size_t)16 * HD_ * sizeof(bf16);
    k_attn<<<dim3(S_ / 16, H_, B_), blk, smem, stream>>>(qRope, kRope, vT, attnB);

    // 5. Output projection (fp32 out straight to d_out).
    k_gemm<<<dim3(D_ / 128, M_ / 128), blk, 0, stream>>>(attnB, WoT, nullptr, nullptr, out, D_, D_);
}
